// LongformerSelfAttention_21354577396082
// MI455X (gfx1250) — compile-verified, matrix operand order checked by reading, not run on hardware
//
#include <hip/hip_runtime.h>
#include <hip/hip_bf16.h>
#include <math.h>

// ---------------- problem constants ----------------
#define BATCH 2
#define SEQ   4096
#define EMB   768
#define HEADS 12
#define HDIM  64
#define WINW  256
#define BS    (BATCH * SEQ)          // 8192 rows total
// attention band per 16-row query tile: [i0-256, i0+288) = 544 keys
#define BAND   544
#define BTILES 34                    // 16-key score tiles
#define KTILES 17                    // 32-key context K-steps
#define VPAD   72                    // padded f16 row pitch of staged V (bank-stagger)

// LDS layout (dynamic): [0, 78336) = V stage (aliases f32 scores), [78336, +17408) = P
#define VST_BYTES (BAND * VPAD * 2)          // 78336
#define P_BYTES   (16 * BAND * 2)            // 17408
#define SMEM_BYTES (VST_BYTES + P_BYTES)     // 95744  (< 320 KB/WGP)

typedef __attribute__((ext_vector_type(16))) _Float16 v16h;
typedef __attribute__((ext_vector_type(8)))  _Float16 v8h;
typedef __attribute__((ext_vector_type(8)))  float    v8f;
typedef __attribute__((ext_vector_type(4)))  int      v4i;

// Address-space-qualified v4i pointers for the async-to-LDS builtin
// (param types revealed by the round-2 diagnostic: v4i AS1*, v4i AS3*, imm, imm).
typedef __attribute__((address_space(1))) v4i* gv4i_p;
typedef __attribute__((address_space(3))) v4i* lv4i_p;

#if defined(__has_builtin)
#if __has_builtin(__builtin_amdgcn_global_load_async_to_lds_b128)
#define HAVE_ASYNC_LDS 1
#endif
#endif
#ifndef HAVE_ASYNC_LDS
#define HAVE_ASYNC_LDS 0
#endif

// WMMA operand fragment from a row-major f16 row pointer (ISA 7.12.2, 16-bit
// A 16x32): lane owns row/col (lane&15), half = lane>>4 selects K sub-range;
// VGPR j holds K = (j>=4?16:0) + half*8 + (j&3)*2 (+1). Contiguous per lane,
// so the compiler merges these into two global_load_b128 / ds_load_b128.
__device__ __forceinline__ v16h frag_row(const _Float16* __restrict__ rowp, int half) {
  v16h a;
#pragma unroll
  for (int j = 0; j < 8; ++j) {
    int kk = ((j >> 2) << 4) + (half << 3) + ((j & 3) << 1);
    a[2 * j]     = rowp[kk];
    a[2 * j + 1] = rowp[kk + 1];
  }
  return a;
}

#define WMMA_F16(A, B, C) \
  __builtin_amdgcn_wmma_f32_16x16x32_f16(false, (A), false, (B), (short)0, (C), false, false)

// ---------------- f32 -> f16 conversion ----------------
__global__ void cvt_f32_f16(const float* __restrict__ in, _Float16* __restrict__ out, int n) {
  int i = blockIdx.x * blockDim.x + threadIdx.x;
  if (i < n) out[i] = (_Float16)in[i];
}

// ---------------- QKV projection: Y = (X @ W^T + b) * scale ----------------
// Register-blocked: each wave owns a 32x64 output tile (2 A-frags x 4 B-frags
// -> 8 WMMAs per 32-wide K step; 192 WMMAs per wave). B[k][n] = W[n][k] is
// row-contiguous in W, so no transpose staging is needed.
// grid (EMB/64, BS/(32*8)), block 256 (8 waves).
__global__ __launch_bounds__(256) void qkv_gemm(
    const _Float16* __restrict__ X, const _Float16* __restrict__ W,
    const float* __restrict__ bias, _Float16* __restrict__ Y, float scale) {
  int lane = threadIdx.x & 31;
  int wave = threadIdx.x >> 5;
  int row0 = (blockIdx.y * 8 + wave) * 32;
  int col0 = blockIdx.x * 64;
  int mn  = lane & 15;
  int hlf = lane >> 4;

  const _Float16* x0 = X + (size_t)(row0 + mn) * EMB;        // A rows, m-tile 0
  const _Float16* x1 = x0 + (size_t)16 * EMB;                // A rows, m-tile 1
  const _Float16* w0 = W + (size_t)(col0 + mn) * EMB;        // B cols, n-tile 0

  v8f acc[2][4] = {};
  for (int kt = 0; kt < EMB / 32; ++kt) {
    int ko = kt * 32;
    v16h a0 = frag_row(x0 + ko, hlf);
    v16h a1 = frag_row(x1 + ko, hlf);
    v16h b0 = frag_row(w0 + ko, hlf);
    v16h b1 = frag_row(w0 + (size_t)16 * EMB + ko, hlf);
    v16h b2 = frag_row(w0 + (size_t)32 * EMB + ko, hlf);
    v16h b3 = frag_row(w0 + (size_t)48 * EMB + ko, hlf);
    acc[0][0] = WMMA_F16(a0, b0, acc[0][0]);
    acc[0][1] = WMMA_F16(a0, b1, acc[0][1]);
    acc[0][2] = WMMA_F16(a0, b2, acc[0][2]);
    acc[0][3] = WMMA_F16(a0, b3, acc[0][3]);
    acc[1][0] = WMMA_F16(a1, b0, acc[1][0]);
    acc[1][1] = WMMA_F16(a1, b1, acc[1][1]);
    acc[1][2] = WMMA_F16(a1, b2, acc[1][2]);
    acc[1][3] = WMMA_F16(a1, b3, acc[1][3]);
  }

#pragma unroll
  for (int mt = 0; mt < 2; ++mt) {
#pragma unroll
    for (int nt = 0; nt < 4; ++nt) {
      int n = col0 + nt * 16 + mn;
      float bv = bias[n];
#pragma unroll
      for (int r = 0; r < 8; ++r) {
        int row = row0 + mt * 16 + r + 8 * hlf;   // C/D layout: m = r + 8*half
        Y[(size_t)row * EMB + n] = (_Float16)((acc[mt][nt][r] + bv) * scale);
      }
    }
  }
}

// ---------------- sliding-window attention ----------------
// grid (SEQ/16, HEADS, BATCH), block = 1 wave. Scores -> LDS (f32), softmax
// (2 lanes/row + shfl_xor combine), P -> LDS (f16), V band staged to LDS
// (async-to-LDS when available), context via WMMA from LDS fragments.
__global__ __launch_bounds__(32) void longformer_attn(
    const _Float16* __restrict__ Q, const _Float16* __restrict__ K,
    const _Float16* __restrict__ V, const float* __restrict__ amask,
    float* __restrict__ out) {
  extern __shared__ char smem[];
  float*    sc  = (float*)smem;                    // [16][BAND], dead after softmax
  _Float16* Vst = (_Float16*)smem;                 // [BAND][VPAD], aliases sc
  _Float16* P   = (_Float16*)(smem + VST_BYTES);   // [16][BAND]

  int lane = threadIdx.x;
  int nn   = lane & 15;
  int hlf  = lane >> 4;
  int b = blockIdx.z, h = blockIdx.y, t = blockIdx.x;
  int i0 = t * 16;
  int kstart = i0 - WINW;
  size_t bb = (size_t)b * SEQ;

  const float* amrow = amask + bb;
  const _Float16* qrow = Q + (bb + i0 + nn) * EMB + h * HDIM;
  v16h qa0 = frag_row(qrow, hlf);        // d = 0..31
  v16h qa1 = frag_row(qrow + 32, hlf);   // d = 32..63

  // ---- pass 1: banded scores into LDS ----
  for (int kt = 0; kt < BTILES; ++kt) {
    int key = kstart + kt * 16 + nn;     // this lane's key row (B column n)
    bool okk = (key >= 0) && (key < SEQ);
    v16h kb0 = {}, kb1 = {};
    if (okk) {
      const _Float16* krow = K + (bb + key) * EMB + h * HDIM;
      kb0 = frag_row(krow, hlf);
      kb1 = frag_row(krow + 32, hlf);
    }
    v8f acc = {};
    acc = WMMA_F16(qa0, kb0, acc);
    acc = WMMA_F16(qa1, kb1, acc);
#pragma unroll
    for (int r = 0; r < 8; ++r) {
      int m = r + 8 * hlf;
      int i = i0 + m;
      float val;
      if (okk && key >= i - WINW && key <= i + WINW)
        val = acc[r] + ((amrow[key] != 0.0f) ? -10000.0f : 0.0f);
      else
        val = -1e30f;                    // outside band / sequence
      sc[m * BAND + kt * 16 + nn] = val;
    }
  }
  __syncthreads();

  // ---- softmax: two lanes per row, combine halves via shfl_xor(16) ----
  {
    int rm = lane & 15;
    int c0 = hlf * (BAND / 2);
    float mx = -1e30f;
    for (int c = 0; c < BAND / 2; ++c) mx = fmaxf(mx, sc[rm * BAND + c0 + c]);
    mx = fmaxf(mx, __shfl_xor(mx, 16, 32));
    float sum = 0.0f;
    for (int c = 0; c < BAND / 2; ++c) {
      float e = __expf(sc[rm * BAND + c0 + c] - mx);
      sum += e;
      sc[rm * BAND + c0 + c] = e;
    }
    sum += __shfl_xor(sum, 16, 32);
    float inv = 1.0f / sum;
    for (int c = 0; c < BAND / 2; ++c)
      P[rm * BAND + c0 + c] = (_Float16)(sc[rm * BAND + c0 + c] * inv);
  }
  __syncthreads();   // sc dead; its LDS is reused as the V stage below

  // ---- stage the 544x64 V band into LDS (coalesced b128 per lane) ----
  for (int it = 0; it < (BAND * 8) / 32; ++it) {     // 136 iterations
    int idx  = it * 32 + lane;
    int krow = idx >> 3;                 // 0..543
    int ch   = idx & 7;                  // 8 f16 per chunk
    int key  = kstart + krow;
    _Float16* ldst = Vst + (size_t)krow * VPAD + ch * 8;
    if (key >= 0 && key < SEQ) {
      const _Float16* gsrc = V + (bb + key) * EMB + h * HDIM + ch * 8;
#if HAVE_ASYNC_LDS
      __builtin_amdgcn_global_load_async_to_lds_b128(
          (gv4i_p)(uintptr_t)gsrc, (lv4i_p)(uintptr_t)ldst, 0, 0);
#else
      *(v8h*)ldst = *(const v8h*)gsrc;
#endif
    } else {
      v8h z = {};
      *(v8h*)ldst = z;                   // zero-fill; P is 0 there anyway
    }
  }
#if HAVE_ASYNC_LDS
  asm volatile("s_wait_asynccnt 0" ::: "memory");
#endif
  __syncthreads();

  // ---- pass 2: context = P (16x544) @ Vst (544x64), all operands from LDS ----
  v8f accO[4] = {};
  for (int kt2 = 0; kt2 < KTILES; ++kt2) {
    v16h pa = frag_row(P + (size_t)(lane & 15) * BAND + kt2 * 32, hlf);
#pragma unroll
    for (int nt = 0; nt < 4; ++nt) {
      v16h vb;
#pragma unroll
      for (int j = 0; j < 8; ++j) {
        int kk = ((j >> 2) << 4) + (hlf << 3) + ((j & 3) << 1);
        int k0 = kt2 * 32 + kk;
        vb[2 * j]     = Vst[(size_t)k0 * VPAD + nt * 16 + nn];
        vb[2 * j + 1] = Vst[(size_t)(k0 + 1) * VPAD + nt * 16 + nn];
      }
      accO[nt] = WMMA_F16(pa, vb, accO[nt]);
    }
  }
#pragma unroll
  for (int nt = 0; nt < 4; ++nt) {
#pragma unroll
    for (int r = 0; r < 8; ++r) {
      int row = i0 + r + 8 * hlf;
      out[(bb + row) * EMB + h * HDIM + nt * 16 + nn] = accO[nt][r];
    }
  }
}

// ---------------- launch ----------------
extern "C" void kernel_launch(void* const* d_in, const int* in_sizes, int n_in,
                              void* d_out, int out_size, void* d_ws, size_t ws_size,
                              hipStream_t stream) {
  const float* hs = (const float*)d_in[0];
  const float* am = (const float*)d_in[1];
  const float* Wq = (const float*)d_in[2];
  const float* bq = (const float*)d_in[3];
  const float* Wk = (const float*)d_in[4];
  const float* bk = (const float*)d_in[5];
  const float* Wv = (const float*)d_in[6];
  const float* bv = (const float*)d_in[7];
  float* out = (float*)d_out;

  const size_t nHS = (size_t)BS * EMB;       // 6,291,456
  const size_t nW  = (size_t)EMB * EMB;      //   589,824

  char* ws = (char*)d_ws;
  _Float16* hs16 = (_Float16*)ws;                 ws += nHS * 2;
  _Float16* wq16 = (_Float16*)ws;                 ws += nW * 2;
  _Float16* wk16 = (_Float16*)ws;                 ws += nW * 2;
  _Float16* wv16 = (_Float16*)ws;                 ws += nW * 2;
  _Float16* q16  = (_Float16*)ws;                 ws += nHS * 2;
  _Float16* k16  = (_Float16*)ws;                 ws += nHS * 2;
  _Float16* v16p = (_Float16*)ws;

  cvt_f32_f16<<<(int)((nHS + 255) / 256), 256, 0, stream>>>(hs, hs16, (int)nHS);
  cvt_f32_f16<<<(int)((nW + 255) / 256), 256, 0, stream>>>(Wq, wq16, (int)nW);
  cvt_f32_f16<<<(int)((nW + 255) / 256), 256, 0, stream>>>(Wk, wk16, (int)nW);
  cvt_f32_f16<<<(int)((nW + 255) / 256), 256, 0, stream>>>(Wv, wv16, (int)nW);

  dim3 ggrid(EMB / 64, BS / (32 * 8));  // (12, 32), 8 waves/block, 32x64 per wave
  qkv_gemm<<<ggrid, 256, 0, stream>>>(hs16, wq16, bq, q16, 0.125f); // 1/sqrt(64)
  qkv_gemm<<<ggrid, 256, 0, stream>>>(hs16, wk16, bk, k16, 1.0f);
  qkv_gemm<<<ggrid, 256, 0, stream>>>(hs16, wv16, bv, v16p, 1.0f);

  dim3 agrid(SEQ / 16, HEADS, BATCH);   // (256, 12, 2)
  longformer_attn<<<agrid, 32, SMEM_BYTES, stream>>>(q16, k16, v16p, am, out);
}
